// Policy_70428873720506
// MI455X (gfx1250) — matrix operand order repaired and checked by reading, not compiled, not run on hardware
//
#include <hip/hip_runtime.h>

typedef __attribute__((ext_vector_type(16))) _Float16 v16h;
typedef __attribute__((ext_vector_type(8)))  _Float16 v8h;
typedef __attribute__((ext_vector_type(8)))  float    v8f;

#define SEQ_LEN 40
// DT*TAU_MEM_INV = 0.1 ; (1 - DT*TAU_SYN_INV) = 0.8

__global__ __launch_bounds__(128, 1)
void snn_policy_wmma(const float* __restrict__ x,
                     const float* __restrict__ w_in,
                     const float* __restrict__ w_rec,
                     const float* __restrict__ w_out,
                     float* __restrict__ out)
{
    // LDS: f16 weights + per-wave z transpose scratch
    __shared__ _Float16 sWrec[128 * 128];   // 32 KB, row n (hidden-out), col k (hidden-in)
    __shared__ _Float16 sWin [128 * 32];    //  8 KB, row h, col k (k<8 real, rest 0)
    __shared__ _Float16 sWout[16 * 128];    //  4 KB, row n (o<2 real, rest 0), col k=h
    __shared__ _Float16 sZ   [4][16 * 128]; // 16 KB, per-wave z_new scratch (m, h)

    const int tid    = threadIdx.x;
    const int lane   = tid & 31;
    const int wv     = tid >> 5;
    const int halfId = lane >> 4;   // 0: lanes 0-15, 1: lanes 16-31
    const int ln     = lane & 15;

    // ---- stage weights into LDS as f16 ----
    for (int idx = tid; idx < 128 * 128; idx += 128)
        sWrec[idx] = (_Float16)w_rec[idx];
    for (int idx = tid; idx < 128 * 32; idx += 128) {
        int row = idx >> 5, k = idx & 31;
        sWin[idx] = (k < 8) ? (_Float16)w_in[row * 8 + k] : (_Float16)0.0f;
    }
    for (int idx = tid; idx < 16 * 128; idx += 128) {
        int n = idx >> 7, k = idx & 127;
        sWout[idx] = (n < 2) ? (_Float16)w_out[n * 128 + k] : (_Float16)0.0f;
    }
    __syncthreads();

    _Float16* zrow = &sZ[wv][0];
    const int batch_base = (int)blockIdx.x * 64 + wv * 16;

    // ---- encoder: keep only 50*x, recompute +/- relu currents per step ----
    float x50[4], venc[8];
    {
        const float* xr = x + (size_t)(batch_base + ln) * 4;
        x50[0] = 50.f * xr[0]; x50[1] = 50.f * xr[1];
        x50[2] = 50.f * xr[2]; x50[3] = 50.f * xr[3];
    }
    #pragma unroll
    for (int f = 0; f < 8; ++f) venc[f] = 0.f;

    // recurrent state in WMMA C/D layout: element r of tile t is (m = r + 8*halfId, h = 16t + ln)
    v8f iacc[8], vmem[8];
    #pragma unroll
    for (int t = 0; t < 8; ++t) { iacc[t] = (v8f)0.f; vmem[t] = (v8f)0.f; }

    // readout state tiles (C layout, lane-columns 0/1 = the two logits)
    v8f vo = (v8f)0.f, io = (v8f)0.f, vm = (v8f)(-__builtin_inff());

    // z A-fragments carried in registers across steps (z0 = 0)
    v16h zA[4];
    #pragma unroll
    for (int kt = 0; kt < 4; ++kt) zA[kt] = (v16h)(_Float16)0.0f;

    for (int step = 0; step < SEQ_LEN; ++step) {
        // 1) LIF encoder step -> spike A-fragment (only K=0..7 nonzero => lanes 0-15, regs 0-3)
        v16h spkA;
        {
            union { v16h v; _Float16 e[16]; } u;
            #pragma unroll
            for (int f = 0; f < 8; ++f) {
                float c  = (f < 4) ? fmaxf(0.f, x50[f]) : fmaxf(0.f, -x50[f - 4]);
                float vn = venc[f] + 0.1f * (c - venc[f]);
                bool sp  = vn > 1.0f;
                venc[f]  = sp ? 0.0f : vn;
                u.e[f]   = (sp && halfId == 0) ? (_Float16)1.0f : (_Float16)0.0f;
            }
            #pragma unroll
            for (int f = 8; f < 16; ++f) u.e[f] = (_Float16)0.0f;
            spkA = u.v;
        }

        // 2) membrane: v_dec = v + 0.1*(i - v); spike; reset; scatter z_new (f16) to LDS
        #pragma unroll
        for (int t = 0; t < 8; ++t) {
            v8f vt = vmem[t];
            v8f it = iacc[t];
            #pragma unroll
            for (int r = 0; r < 8; ++r) {
                float vd = vt[r] + 0.1f * (it[r] - vt[r]);
                bool sp = vd > 1.0f;
                vt[r] = sp ? 0.0f : vd;
                zrow[(r + halfId * 8) * 128 + t * 16 + ln] =
                    sp ? (_Float16)1.0f : (_Float16)0.0f;
            }
            vmem[t] = vt;
        }

        // 3) synaptic current: i_new = 0.8*i + spk@w_in^T + z_old@w_rec^T
        //    (zA still holds z_old in registers; the LDS scatter above doesn't touch it)
        #pragma unroll
        for (int t = 0; t < 8; ++t) {
            v8f acc = iacc[t] * 0.8f;
            const v16h bIn = *(const v16h*)&sWin[(t * 16 + ln) * 32 + halfId * 16];
            acc = __builtin_amdgcn_wmma_f32_16x16x32_f16(false, spkA, false, bIn,
                                                         (short)0, acc, false, false);
            #pragma unroll
            for (int kt = 0; kt < 4; ++kt) {
                const v16h bR = *(const v16h*)&sWrec[(t * 16 + ln) * 128 + kt * 32 + halfId * 16];
                acc = __builtin_amdgcn_wmma_f32_16x16x32_f16(false, zA[kt], false, bR,
                                                             (short)0, acc, false, false);
            }
            iacc[t] = acc;
        }

        // 4) reload z_new as A-fragments (serves the readout now, and is z_old next step)
        #pragma unroll
        for (int kt = 0; kt < 4; ++kt) {
            int k0 = kt * 32 + halfId * 16;
            union { v16h v; v8h h[2]; } u;
            u.h[0] = *(const v8h*)&zrow[ln * 128 + k0];
            u.h[1] = *(const v8h*)&zrow[ln * 128 + k0 + 8];
            zA[kt] = u.v;
        }

        // 5) LI readout on the tensor path: io_new = 0.8*io + z_new @ w_out^T
        //    (vo_new must use old io, so update vo first; then fuse 0.8*io as WMMA C)
        {
            vo = vo + 0.1f * (io - vo);
            v8f s = io * 0.8f;
            #pragma unroll
            for (int kt = 0; kt < 4; ++kt) {
                const v16h bO = *(const v16h*)&sWout[ln * 128 + kt * 32 + halfId * 16];
                s = __builtin_amdgcn_wmma_f32_16x16x32_f16(false, zA[kt], false, bO,
                                                           (short)0, s, false, false);
            }
            io = s;
            #pragma unroll
            for (int r = 0; r < 8; ++r) vm[r] = fmaxf(vm[r], vo[r]);
        }
    }

    // 6) softmax over the 2 logits; lane-column 0 of each half writes its 8 rows
    #pragma unroll
    for (int r = 0; r < 8; ++r) {
        float other = __shfl_xor(vm[r], 1, 32);   // swap logit columns (lane 0 <-> 1)
        if (ln == 0) {
            int row = batch_base + halfId * 8 + r;
            float a = vm[r], b = other;
            float mx = fmaxf(a, b);
            float ea = __expf(a - mx), eb = __expf(b - mx);
            float inv = 1.0f / (ea + eb);
            out[row * 2 + 0] = ea * inv;
            out[row * 2 + 1] = eb * inv;
        }
    }
}

extern "C" void kernel_launch(void* const* d_in, const int* in_sizes, int n_in,
                              void* d_out, int out_size, void* d_ws, size_t ws_size,
                              hipStream_t stream) {
    const float* x     = (const float*)d_in[0];
    const float* w_in  = (const float*)d_in[1];
    const float* w_rec = (const float*)d_in[2];
    const float* w_out = (const float*)d_in[3];
    float* out = (float*)d_out;
    int batch  = in_sizes[0] / 4;        // x is (B, 4)
    int blocks = batch / 64;             // 4 waves/block * 16 rows/wave
    snn_policy_wmma<<<blocks, 128, 0, stream>>>(x, w_in, w_rec, w_out, out);
}
